// Model_61959198212618
// MI455X (gfx1250) — matrix operand order repaired and checked by reading, not compiled, not run on hardware
//
#include <hip/hip_runtime.h>
#include <hip/hip_bf16.h>

// Problem dims (from reference): B=4, T=8, C=32, H=W=40, K=9
#define BB 4
#define TT 8
#define CC 32
#define HH 40
#define WW 40
#define HW 1600          // H*W nodes per frame
#define KK 9             // edges per node
#define NPAIR 28         // B*(T-1)
#define EPS 1e-8f

typedef _Float16 v16h __attribute__((ext_vector_type(16)));
typedef _Float16 v8h  __attribute__((ext_vector_type(8)));
typedef float    v8f  __attribute__((ext_vector_type(8)));

// ---------------------------------------------------------------------------
// Kernel 1: per-node L2 normalization + pack to f16, layout [B*T][HW][32]
// Input sim_feat is [B,T,C,H,W]; for node i of frame bt, channel c lives at
// (bt*C + c)*HW + i  (stride HW between channels).
// ---------------------------------------------------------------------------
__global__ void __launch_bounds__(256)
normalize_pack_kernel(const float* __restrict__ x, _Float16* __restrict__ fn) {
    int g = blockIdx.x * blockDim.x + threadIdx.x;
    if (g >= BB * TT * HW) return;
    int i  = g % HW;
    int bt = g / HW;
    const float* src = x + (size_t)bt * CC * HW + i;
    float v[CC];
    float s = 0.0f;
#pragma unroll
    for (int c = 0; c < CC; ++c) {
        v[c] = src[(size_t)c * HW];
        s += v[c] * v[c];
    }
    float r = rsqrtf(s + EPS);
    _Float16* dst = fn + ((size_t)bt * HW + i) * CC;
#pragma unroll
    for (int c = 0; c < CC; ++c) dst[c] = (_Float16)(v[c] * r);
}

// ---------------------------------------------------------------------------
// Kernel 2: fused  sim-tile WMMA  ->  LDS slab  ->  streaming top-9  ->
//           weight-normalize + displacement init.
// One wave (32 threads) handles 16 source rows of one (b,t) pair.
// grid = (HW/16 = 100, NPAIR = 28), block = 32.
//
// A fragment (16x32 f16, contiguous K): lane l (m=l&15, h=l>>4) holds row m,
//   K = 16h..16h+15  -> two contiguous 8-half loads.
// B fragment (32x16 f16): lane l (n=l&15, h=l>>4) holds column-node n,
//   K = 16h..16h+15 -> one contiguous 16-half load.
// D (16x16 f32): vgpr r, lane l -> row (r + 8h), col n.
// ---------------------------------------------------------------------------
#define CHUNK_TILES 25                  // 25 tiles * 16 = 400 columns per chunk
#define CHUNK_COLS  (CHUNK_TILES * 16)

__global__ void __launch_bounds__(32)
sim_topk_kernel(const _Float16* __restrict__ fn,
                float* __restrict__ outW,
                int*   __restrict__ outI,
                int*   __restrict__ outNI) {
    __shared__ float smem[16 * CHUNK_COLS];   // 25.6 KB: 16 rows x 400 cols

    const int lane    = threadIdx.x;
    const int p       = blockIdx.y;          // pair index 0..27
    const int b       = p / (TT - 1);
    const int t       = p % (TT - 1);
    const int tileRow = blockIdx.x;          // 0..99

    const _Float16* fA = fn + ((size_t)(b * TT + t)     * HW) * CC; // frame t
    const _Float16* fB = fn + ((size_t)(b * TT + t + 1) * HW) * CC; // frame t+1

    const int m = lane & 15;
    const int h = lane >> 4;

    // ---- A fragment: loaded once per block (invariant over column tiles) ----
    const _Float16* abase = fA + (size_t)(tileRow * 16 + m) * CC;
    v8h alo = *(const v8h*)(abase + 16 * h);        // K = 16h .. 16h+7
    v8h ahi = *(const v8h*)(abase + 16 * h + 8);   // K = 16h+8 .. 16h+15
    v16h afrag = __builtin_shufflevector(alo, ahi,
        0, 1, 2, 3, 4, 5, 6, 7, 8, 9, 10, 11, 12, 13, 14, 15);

    // ---- running top-9 (sorted descending), per source row (lanes 0..15) ----
    float tv[KK];
    int   ti[KK];
#pragma unroll
    for (int k = 0; k < KK; ++k) { tv[k] = -3.4e38f; ti[k] = 0; }

    for (int chunk = 0; chunk < 4; ++chunk) {
        // Produce 400 sim columns into LDS via 25 WMMAs.
        for (int jt = 0; jt < CHUNK_TILES; ++jt) {
            const int j = chunk * CHUNK_TILES + jt;
            const _Float16* bbase =
                fB + (size_t)(j * 16 + m) * CC + 16 * h;   // col-node m, K=16h..
            v16h bfrag = *(const v16h*)bbase;

            v8f acc = {};
            acc = __builtin_amdgcn_wmma_f32_16x16x32_f16(
                /*neg_a=*/false, afrag, /*neg_b=*/false, bfrag,
                /*c_mod=*/(short)0, acc, /*reuse_a=*/false, /*reuse_b=*/false);

#pragma unroll
            for (int r = 0; r < 8; ++r)
                smem[(r + 8 * h) * CHUNK_COLS + jt * 16 + m] = acc[r];
        }
        __syncthreads();

        // Lanes 0..15: stream-scan this chunk's 400 columns for row `lane`.
        if (lane < 16) {
            const float* row = smem + lane * CHUNK_COLS;
            for (int c = 0; c < CHUNK_COLS; ++c) {
                float v = row[c];
                if (v > tv[KK - 1]) {              // strict > : ties keep lower idx
                    tv[KK - 1] = v;
                    ti[KK - 1] = chunk * CHUNK_COLS + c;
#pragma unroll
                    for (int k = KK - 1; k > 0; --k) {
                        if (tv[k] > tv[k - 1]) {   // strict > : stable ordering
                            float fv = tv[k]; tv[k] = tv[k - 1]; tv[k - 1] = fv;
                            int   iv = ti[k]; ti[k] = ti[k - 1]; ti[k - 1] = iv;
                        }
                    }
                }
            }
        }
        __syncthreads();
    }

    // ---- edge-normalize + coordinate displacement, write outputs ----
    if (lane < 16) {
        const int row = tileRow * 16 + lane;
        float s = 0.0f;
#pragma unroll
        for (int k = 0; k < KK; ++k) s += tv[k];
        const float inv = 1.0f / (s + EPS);
        const int sy = row / WW;
        const int sx = row % WW;
        const size_t base = ((size_t)p * HW + row) * KK;
#pragma unroll
        for (int k = 0; k < KK; ++k) {
            const size_t pos = base + k;
            outW[pos]        = tv[k] * inv;
            outI[pos]        = ti[k];
            outNI[2 * pos]     = ti[k] / WW - sy;   // ty - sy
            outNI[2 * pos + 1] = ti[k] % WW - sx;   // tx - sx
        }
    }
}

// ---------------------------------------------------------------------------
// Host-side launcher. d_out layout (concatenated, return order):
//   wgt       : 28*1600*9       = 403200 f32   [offset 0]
//   idx       : 28*1600*9       = 403200 i32   [offset 403200]
//   node_init : 28*1600*9*2     = 806400 i32   [offset 806400]
// d_ws: packed normalized f16 features, 4*8*1600*32 halves = 3,276,800 bytes.
// ---------------------------------------------------------------------------
extern "C" void kernel_launch(void* const* d_in, const int* in_sizes, int n_in,
                              void* d_out, int out_size, void* d_ws, size_t ws_size,
                              hipStream_t stream) {
    const float* sim_feat = (const float*)d_in[0];
    _Float16*    fn       = (_Float16*)d_ws;

    float* outW  = (float*)d_out;
    int*   outI  = (int*)d_out + 403200;
    int*   outNI = (int*)d_out + 806400;

    // Kernel 1: 51200 nodes, 256 threads/block.
    const int nNodes = BB * TT * HW;
    normalize_pack_kernel<<<(nNodes + 255) / 256, 256, 0, stream>>>(sim_feat, fn);

    // Kernel 2: (100 row-tiles) x (28 pairs), one wave each.
    sim_topk_kernel<<<dim3(HW / 16, NPAIR), 32, 0, stream>>>(fn, outW, outI, outNI);
}
